// QuantumOCTClassifier_61340722921807
// MI455X (gfx1250) — compile-verified, matrix operand order checked by reading, not run on hardware
//
#include <hip/hip_runtime.h>
#include <hip/hip_bf16.h>

typedef __attribute__((ext_vector_type(16))) _Float16 v16h;
typedef __attribute__((ext_vector_type(8)))  float    v8f;

#define NQ 8
#define NL 8
#define BATCH 4096
#define BN_EPS 1e-5f

// ---------------------------------------------------------------------------
// WMMA helpers (gfx1250 16x16x32 f16 layout, CDNA5 ISA §7.12.2)
// A operand (16xK, lane = M) / B operand (Kx16, lane = N), per lane 16 halves:
//   element e -> r = e>>1, t = e&1 ; K = 2r + t + 8*hi + (r>=4 ? 8 : 0)
// C/D: VGPR v -> M = v + 8*hi ; N = lane & 15
// ---------------------------------------------------------------------------
__device__ __forceinline__ int wmma_k(int e, int hi) {
    int r = e >> 1, t = e & 1;
    return 2 * r + t + 8 * hi + ((r >= 4) ? 8 : 0);
}

__device__ __forceinline__ v8f wmma_f16(v16h a, v16h b, v8f c) {
    return __builtin_amdgcn_wmma_f32_16x16x32_f16(false, a, false, b,
                                                  (short)0, c, false, false);
}

// ===========================================================================
// Kernel 0: precompute trig table for the (sample-independent) layer weights.
// trig[(l*8+q)*4 + {0,1,2,3}] = { cos(qw0/2), sin(qw0/2), cos(qw1/2), sin(qw1/2) }
// ===========================================================================
__global__ void __launch_bounds__(256)
trig_kernel(const float* __restrict__ qw, float* __restrict__ trig) {
    const int id = threadIdx.x;                 // 0..255, single block
    const int lq = id >> 2, sel = id & 3;
    const float v = 0.5f * qw[lq * 2 + (sel >> 1)];
    trig[id] = (sel & 1) ? sinf(v) : cosf(v);
}

// ===========================================================================
// Kernel 1: h8 = tanh( BN(relu(x @ w1.T + b1)) @ w2.T + b2 )   (4096 x 8)
// 1 wave per 16 rows; 4 waves / block.
// ===========================================================================
__global__ void __launch_bounds__(128)
head_kernel(const float* __restrict__ x,  const float* __restrict__ w1,
            const float* __restrict__ b1, const float* __restrict__ g1,
            const float* __restrict__ be1,const float* __restrict__ m1,
            const float* __restrict__ v1, const float* __restrict__ w2,
            const float* __restrict__ b2, float* __restrict__ h8) {
    __shared__ float lds[4][16 * 32];
    const int lane = threadIdx.x & 31;
    const int wv   = threadIdx.x >> 5;
    const int m0   = (blockIdx.x * 4 + wv) * 16;
    const int idx  = lane & 15, hi = lane >> 4;

    // A: x rows m0..m0+15, K = 64 -> two 16x32 f16 tiles
    v16h a0, a1;
    const float* xr = x + (m0 + idx) * 64;
#pragma unroll
    for (int e = 0; e < 16; ++e) {
        int k = wmma_k(e, hi);
        a0[e] = (_Float16)xr[k];
        a1[e] = (_Float16)xr[32 + k];
    }
    float* hl = lds[wv];
#pragma unroll
    for (int nt = 0; nt < 2; ++nt) {
        const int n0 = nt * 16;
        v16h bq0, bq1;
        const float* wr = w1 + (n0 + idx) * 64;   // w1: (32,64) row-major
#pragma unroll
        for (int e = 0; e < 16; ++e) {
            int k = wmma_k(e, hi);
            bq0[e] = (_Float16)wr[k];
            bq1[e] = (_Float16)wr[32 + k];
        }
        v8f acc = {};
        acc = wmma_f16(a0, bq0, acc);
        acc = wmma_f16(a1, bq1, acc);
        const int c   = n0 + idx;
        const float bias  = b1[c];
        const float scale = g1[c] * rsqrtf(v1[c] + BN_EPS);
        const float shift = be1[c] - m1[c] * scale;
#pragma unroll
        for (int v = 0; v < 8; ++v) {
            float hv = fmaxf(acc[v] + bias, 0.0f);           // relu
            hl[(v + 8 * hi) * 32 + c] = hv * scale + shift;  // BN (eval)
        }
    }
    __syncthreads();

    // h1(16x32) @ w2.T(32x8, N padded to 16)
    v16h a2, b2v;
#pragma unroll
    for (int e = 0; e < 16; ++e) {
        int k = wmma_k(e, hi);
        a2[e]  = (_Float16)hl[idx * 32 + k];
        b2v[e] = (idx < 8) ? (_Float16)w2[idx * 32 + k] : (_Float16)0.0f;
    }
    v8f acc = {};
    acc = wmma_f16(a2, b2v, acc);
    if (idx < 8) {
        const float bias = b2[idx];
#pragma unroll
        for (int v = 0; v < 8; ++v)
            h8[(m0 + v + 8 * hi) * 8 + idx] = tanhf(acc[v] + bias);
    }
}

// ===========================================================================
// Kernel 2: 8-qubit statevector simulator, one wave32 per sample.
// Amplitude index a = (lane<<3) | r ; qubit q lives at bit b = 7-q.
//   b < 3  : pairing across registers r, r^(1<<b)   (pure VALU)
//   b >= 3 : pairing across lanes via __shfl_xor    (ds datapath)
// ===========================================================================
template<int B>
__device__ __forceinline__ void gate_rx(float (&re)[8], float (&im)[8],
                                        float c, float s, int lane) {
    // new_self = c*self + (-i s)*other  (symmetric in both positions)
    if constexpr (B < 3) {
#pragma unroll
        for (int r = 0; r < 8; ++r) {
            if (((r >> B) & 1) == 0) {
                const int r1 = r | (1 << B);
                float a0r = re[r], a0i = im[r], a1r = re[r1], a1i = im[r1];
                re[r]  = c * a0r + s * a1i;  im[r]  = c * a0i - s * a1r;
                re[r1] = c * a1r + s * a0i;  im[r1] = c * a1i - s * a0r;
            }
        }
    } else {
        const int msk = 1 << (B - 3);
#pragma unroll
        for (int r = 0; r < 8; ++r) {
            float orr = __shfl_xor(re[r], msk, 32);
            float oii = __shfl_xor(im[r], msk, 32);
            float sr = re[r], si = im[r];
            re[r] = c * sr + s * oii;
            im[r] = c * si - s * orr;
        }
    }
}

template<int B>
__device__ __forceinline__ void gate_ry(float (&re)[8], float (&im)[8],
                                        float cy, float sy, int lane) {
    if constexpr (B < 3) {
#pragma unroll
        for (int r = 0; r < 8; ++r) {
            if (((r >> B) & 1) == 0) {
                const int r1 = r | (1 << B);
                float a0r = re[r], a0i = im[r], a1r = re[r1], a1i = im[r1];
                re[r]  = cy * a0r - sy * a1r;  im[r]  = cy * a0i - sy * a1i;
                re[r1] = sy * a0r + cy * a1r;  im[r1] = sy * a0i + cy * a1i;
            }
        }
    } else {
        const int msk = 1 << (B - 3);
        const float sgn = ((lane >> (B - 3)) & 1) ? sy : -sy;
#pragma unroll
        for (int r = 0; r < 8; ++r) {
            float orr = __shfl_xor(re[r], msk, 32);
            float oii = __shfl_xor(im[r], msk, 32);
            re[r] = cy * re[r] + sgn * orr;
            im[r] = cy * im[r] + sgn * oii;
        }
    }
}

template<int B>
__device__ __forceinline__ void gate_rz(float (&re)[8], float (&im)[8],
                                        float cz, float sz, int lane) {
    // diag(e^{-i t/2}, e^{+i t/2}); phase imag = (bit ? +sz : -sz)
    if constexpr (B < 3) {
#pragma unroll
        for (int r = 0; r < 8; ++r) {
            const float pim = ((r >> B) & 1) ? sz : -sz;
            float rr = re[r], ii = im[r];
            re[r] = cz * rr - pim * ii;
            im[r] = cz * ii + pim * rr;
        }
    } else {
        const float pim = ((lane >> (B - 3)) & 1) ? sz : -sz;
#pragma unroll
        for (int r = 0; r < 8; ++r) {
            float rr = re[r], ii = im[r];
            re[r] = cz * rr - pim * ii;
            im[r] = cz * ii + pim * rr;
        }
    }
}

template<int BC, int BT>
__device__ __forceinline__ void gate_cnot(float (&re)[8], float (&im)[8], int lane) {
    if constexpr (BT < 3) {
        if constexpr (BC < 3) {
#pragma unroll
            for (int r = 0; r < 8; ++r) {
                if (((r >> BC) & 1) && !((r >> BT) & 1)) {
                    const int r1 = r | (1 << BT);
                    float t = re[r]; re[r] = re[r1]; re[r1] = t;
                    t = im[r]; im[r] = im[r1]; im[r1] = t;
                }
            }
        } else {
            const bool ctrl = (lane >> (BC - 3)) & 1;
#pragma unroll
            for (int r = 0; r < 8; ++r) {
                if (((r >> BT) & 1) == 0) {
                    const int r1 = r | (1 << BT);
                    float r0r = re[r], r1r = re[r1], r0i = im[r], r1i = im[r1];
                    re[r] = ctrl ? r1r : r0r;  re[r1] = ctrl ? r0r : r1r;
                    im[r] = ctrl ? r1i : r0i;  im[r1] = ctrl ? r0i : r1i;
                }
            }
        }
    } else {
        const int msk = 1 << (BT - 3);
        if constexpr (BC < 3) {
#pragma unroll
            for (int r = 0; r < 8; ++r) {
                if ((r >> BC) & 1) {   // uniform across lanes: r is compile-time
                    re[r] = __shfl_xor(re[r], msk, 32);
                    im[r] = __shfl_xor(im[r], msk, 32);
                }
            }
        } else {
            const bool ctrl = (lane >> (BC - 3)) & 1;
#pragma unroll
            for (int r = 0; r < 8; ++r) {
                float pr = __shfl_xor(re[r], msk, 32);
                float pi = __shfl_xor(im[r], msk, 32);
                re[r] = ctrl ? pr : re[r];
                im[r] = ctrl ? pi : im[r];
            }
        }
    }
}

__global__ void __launch_bounds__(256)
quantum_kernel(const float* __restrict__ h8, const float* __restrict__ trig,
               float* __restrict__ q8) {
    __shared__ float lds_trig[256] __attribute__((aligned(16)));
    const int tid    = threadIdx.x;
    const int lane   = tid & 31;
    const int sample = (blockIdx.x * blockDim.x + tid) >> 5;

    // Stage the 1KB wave-uniform trig table into LDS with the CDNA5
    // async global->LDS path (ASYNCcnt-tracked), one B32 per thread.
    {
        unsigned lds_off = (unsigned)(uintptr_t)&lds_trig[tid];
        unsigned voff    = (unsigned)(tid * 4);
        asm volatile("global_load_async_to_lds_b32 %0, %1, %2"
                     :: "v"(lds_off), "v"(voff), "s"(trig) : "memory");
        asm volatile("s_wait_asynccnt 0" ::: "memory");
    }
    __syncthreads();
    const float4* t4 = (const float4*)lds_trig;   // [l*8+q] = {cy, sy, cz, sz}

    float re[8], im[8];
#pragma unroll
    for (int r = 0; r < 8; ++r) { re[r] = 0.0f; im[r] = 0.0f; }
    if (lane == 0) re[0] = 1.0f;                  // |00000000>

    float ang[8];
#pragma unroll
    for (int q = 0; q < 8; ++q) ang[q] = h8[sample * 8 + q];

    // AngleEmbedding: RX(x_q) on qubit q  (bit b = 7-q)
#define RXQ(Q) { float s_, c_; sincosf(0.5f * ang[Q], &s_, &c_);                 \
                 gate_rx<7-(Q)>(re, im, c_, s_, lane); }
    RXQ(0) RXQ(1) RXQ(2) RXQ(3) RXQ(4) RXQ(5) RXQ(6) RXQ(7)
#undef RXQ

    for (int l = 0; l < NL; ++l) {
#define ROTQ(Q) { float4 t = t4[l * 8 + (Q)];                                    \
                  gate_ry<7-(Q)>(re, im, t.x, t.y, lane);                        \
                  gate_rz<7-(Q)>(re, im, t.z, t.w, lane); }
        ROTQ(0) ROTQ(1) ROTQ(2) ROTQ(3) ROTQ(4) ROTQ(5) ROTQ(6) ROTQ(7)
#undef ROTQ
        // CNOT ring q -> (q+1)%8 : BC = 7-q, BT = 7-(q+1)%8
        gate_cnot<7,6>(re, im, lane);
        gate_cnot<6,5>(re, im, lane);
        gate_cnot<5,4>(re, im, lane);
        gate_cnot<4,3>(re, im, lane);
        gate_cnot<3,2>(re, im, lane);
        gate_cnot<2,1>(re, im, lane);
        gate_cnot<1,0>(re, im, lane);
        gate_cnot<0,7>(re, im, lane);
    }

    // <Z_q> = sum_a p(a) * (bit_{7-q}(a) ? -1 : +1)
    float p[8], psum = 0.0f;
#pragma unroll
    for (int r = 0; r < 8; ++r) { p[r] = re[r]*re[r] + im[r]*im[r]; psum += p[r]; }

    float ez[8];
#pragma unroll
    for (int q = 0; q < 5; ++q)                   // b = 7-q >= 3 : lane-bit sign
        ez[q] = ((lane >> (4 - q)) & 1) ? -psum : psum;
#pragma unroll
    for (int q = 5; q < 8; ++q) {                 // b = 7-q < 3 : register-bit sign
        const int b = 7 - q;
        float s = 0.0f;
#pragma unroll
        for (int r = 0; r < 8; ++r) s += ((r >> b) & 1) ? -p[r] : p[r];
        ez[q] = s;
    }
#pragma unroll
    for (int q = 0; q < 8; ++q) {
        float v = ez[q];
#pragma unroll
        for (int m = 16; m >= 1; m >>= 1) v += __shfl_xor(v, m, 32);
        ez[q] = v;
    }
    if (lane == 0) {
#pragma unroll
        for (int q = 0; q < 8; ++q) q8[sample * 8 + q] = ez[q];
    }
}

// ===========================================================================
// Kernel 3: tail MLP 8->32 (BN) ->16->1
// ===========================================================================
__global__ void __launch_bounds__(128)
tail_kernel(const float* __restrict__ q8, const float* __restrict__ w3,
            const float* __restrict__ b3, const float* __restrict__ g2,
            const float* __restrict__ be2,const float* __restrict__ m2,
            const float* __restrict__ v2, const float* __restrict__ w4,
            const float* __restrict__ b4, const float* __restrict__ w5,
            const float* __restrict__ b5, float* __restrict__ out) {
    __shared__ float lds[4][16 * 32];
    const int lane = threadIdx.x & 31;
    const int wv   = threadIdx.x >> 5;
    const int m0   = (blockIdx.x * 4 + wv) * 16;
    const int idx  = lane & 15, hi = lane >> 4;

    // A1: q8 rows (16 x 8, K padded to 32 with zeros)
    v16h a1;
    const float* qr = q8 + (m0 + idx) * 8;
#pragma unroll
    for (int e = 0; e < 16; ++e) {
        int k = wmma_k(e, hi);
        a1[e] = (k < 8) ? (_Float16)qr[k] : (_Float16)0.0f;
    }
    float* hl = lds[wv];
#pragma unroll
    for (int nt = 0; nt < 2; ++nt) {
        const int n0 = nt * 16;
        v16h bv;
        const float* wr = w3 + (n0 + idx) * 8;          // w3: (32,8)
#pragma unroll
        for (int e = 0; e < 16; ++e) {
            int k = wmma_k(e, hi);
            bv[e] = (k < 8) ? (_Float16)wr[k] : (_Float16)0.0f;
        }
        v8f acc = {};
        acc = wmma_f16(a1, bv, acc);
        const int c = n0 + idx;
        const float bias  = b3[c];
        const float scale = g2[c] * rsqrtf(v2[c] + BN_EPS);
        const float shift = be2[c] - m2[c] * scale;
#pragma unroll
        for (int v = 0; v < 8; ++v) {
            float hv = fmaxf(acc[v] + bias, 0.0f);
            hl[(v + 8 * hi) * 32 + c] = hv * scale + shift;
        }
    }
    __syncthreads();

    // h4 = relu(h3(16x32) @ w4.T(32x16) + b4)
    v16h a2, b4v;
    const float* wr4 = w4 + idx * 32;                   // w4: (16,32)
#pragma unroll
    for (int e = 0; e < 16; ++e) {
        int k = wmma_k(e, hi);
        a2[e]  = (_Float16)hl[idx * 32 + k];
        b4v[e] = (_Float16)wr4[k];
    }
    v8f acc = {};
    acc = wmma_f16(a2, b4v, acc);
    const float bias4 = b4[idx];
    __syncthreads();
#pragma unroll
    for (int v = 0; v < 8; ++v)
        hl[(v + 8 * hi) * 16 + idx] = fmaxf(acc[v] + bias4, 0.0f);
    __syncthreads();

    // out = h4(16x16, K padded to 32) @ w5.T(16x1, N padded to 16) + b5
    v16h a3, b5v;
#pragma unroll
    for (int e = 0; e < 16; ++e) {
        int k = wmma_k(e, hi);
        a3[e]  = (k < 16) ? (_Float16)hl[idx * 16 + k] : (_Float16)0.0f;
        b5v[e] = (idx == 0 && k < 16) ? (_Float16)w5[k] : (_Float16)0.0f;
    }
    v8f accf = {};
    accf = wmma_f16(a3, b5v, accf);
    if (idx == 0) {
        const float bb = b5[0];
#pragma unroll
        for (int v = 0; v < 8; ++v)
            out[m0 + v + 8 * hi] = accf[v] + bb;
    }
}

// ===========================================================================
extern "C" void kernel_launch(void* const* d_in, const int* in_sizes, int n_in,
                              void* d_out, int out_size, void* d_ws, size_t ws_size,
                              hipStream_t stream) {
    const float* x   = (const float*)d_in[0];
    const float* w1  = (const float*)d_in[1];
    const float* b1  = (const float*)d_in[2];
    const float* g1  = (const float*)d_in[3];
    const float* be1 = (const float*)d_in[4];
    const float* m1  = (const float*)d_in[5];
    const float* v1  = (const float*)d_in[6];
    const float* w2  = (const float*)d_in[7];
    const float* b2  = (const float*)d_in[8];
    const float* qw  = (const float*)d_in[9];
    const float* w3  = (const float*)d_in[10];
    const float* b3  = (const float*)d_in[11];
    const float* g2  = (const float*)d_in[12];
    const float* be2 = (const float*)d_in[13];
    const float* m2  = (const float*)d_in[14];
    const float* v2  = (const float*)d_in[15];
    const float* w4  = (const float*)d_in[16];
    const float* b4  = (const float*)d_in[17];
    const float* w5  = (const float*)d_in[18];
    const float* b5  = (const float*)d_in[19];

    float* out  = (float*)d_out;
    float* h8   = (float*)d_ws;             // 4096 x 8 (tanh output / qnode input)
    float* q8   = h8 + BATCH * NQ;          // 4096 x 8 (expectation values)
    float* trig = q8 + BATCH * NQ;          // 256 floats: per-(l,q) {cy,sy,cz,sz}

    trig_kernel<<<1, 256, 0, stream>>>(qw, trig);
    // 256 waves of 16 rows each; 4 waves per block
    head_kernel<<<BATCH / 64, 128, 0, stream>>>(x, w1, b1, g1, be1, m1, v1,
                                                w2, b2, h8);
    // one wave32 per sample; 8 samples per 256-thread block
    quantum_kernel<<<BATCH / 8, 256, 0, stream>>>(h8, trig, q8);
    tail_kernel<<<BATCH / 64, 128, 0, stream>>>(q8, w3, b3, g2, be2, m2, v2,
                                                w4, b4, w5, b5, out);
}
